// PathAttention_21492016349768
// MI455X (gfx1250) — matrix operand order repaired and checked by reading, not compiled, not run on hardware
//
#include <hip/hip_runtime.h>
#include <hip/hip_bf16.h>

#define NTOK   1024
#define DMODEL 1024
#define NHEAD  16
#define HDIM   64
#define BATCH  2
#define NBH    (BATCH*NHEAD)
#define MROWS  (BATCH*NTOK)
#define ATT_SCALE 0.125f

typedef __attribute__((ext_vector_type(16))) __bf16         v16bf;
typedef __attribute__((ext_vector_type(16))) unsigned short v16u;
typedef __attribute__((ext_vector_type(8)))  float          v8f;
typedef __attribute__((ext_vector_type(4)))  unsigned int   v4u;
typedef __attribute__((ext_vector_type(8)))  int            v8i;
typedef __attribute__((ext_vector_type(4)))  int            v4i;

#define WMMA_BF16(a,b,c) __builtin_amdgcn_wmma_f32_16x16x32_bf16(false,(a),false,(b),(short)0,(c),false,false)

#if defined(__has_builtin)
#if __has_builtin(__builtin_amdgcn_tensor_load_to_lds) && __has_builtin(__builtin_amdgcn_s_wait_tensorcnt)
#define HAVE_TDM 1
#endif
#endif
#ifndef HAVE_TDM
#define HAVE_TDM 0
#endif

// ---- fragment helpers (wave32 CDNA5 WMMA operand layouts) -------------------
static __device__ __forceinline__ v16bf load_a_frag(const __bf16* rowBase, int ld, int kb) {
  int lane = threadIdx.x & 31;
  int r = lane & 15, hf = lane >> 4;
  const __bf16* p0 = rowBase + (size_t)r * ld + kb + hf * 8;
  v16bf a;
  ((uint4*)&a)[0] = *(const uint4*)p0;
  ((uint4*)&a)[1] = *(const uint4*)(p0 + 16);
  return a;
}
static __device__ __forceinline__ v16bf load_b_frag(const __bf16* bt, int ld, int col0, int kb) {
  int lane = threadIdx.x & 31;
  int c = lane & 15, hf = lane >> 4;
  const __bf16* p = bt + (size_t)(col0 + c) * ld + kb + hf * 8;
  v16bf b;
  ((uint4*)&b)[0] = *(const uint4*)p;
  ((uint4*)&b)[1] = *(const uint4*)(p + 16);
  return b;
}
static __device__ __forceinline__ v16bf neg_frag(v16bf a) {
  v16u u = __builtin_bit_cast(v16u, a);
  u ^= (unsigned short)0x8000u;
  return __builtin_bit_cast(v16bf, u);
}

#if HAVE_TDM
// Issue a TDM 2D tile load: 128 rows x 32 bf16, row stride = strideElems,
// into LDS with +16B pad after every 64B row (20-bank stride, 16B aligned).
// D# layout per CDNA5 ISA.
static __device__ __forceinline__ void tdm_issue_2d(unsigned ldsOff, const __bf16* g, int strideElems) {
  unsigned long long ga = (unsigned long long)(size_t)g;
  v4u g0;
  g0.x = 1u;                                             // count=1, user descriptor
  g0.y = ldsOff;                                         // lds_addr (bytes)
  g0.z = (unsigned)ga;                                   // global_addr[31:0]
  g0.w = ((unsigned)(ga >> 32) & 0x01ffffffu) | 0x80000000u; // addr[56:32] | type=2
  union { unsigned w[8]; v8i v; } g1 = {};
  g1.w[0] = (1u << 16)      // data_size = 2 bytes
          | (1u << 20)      // pad_enable
          | (3u << 22)      // pad_interval: every 16 DWORDs (64B)
          | (3u << 25);     // pad_amount: 4 DWORDs (16B)
  g1.w[1] = 32u << 16;                 // tensor_dim0[15:0] = 32
  g1.w[2] = (128u << 16);              // tensor_dim0 hi=0 | tensor_dim1[15:0]=128
  g1.w[3] = (32u << 16);               // tensor_dim1 hi=0 | tile_dim0=32
  g1.w[4] = 128u;                      // tile_dim1=128, tile_dim2=0
  g1.w[5] = (unsigned)strideElems;     // tensor_dim0_stride[31:0]
  g1.w[6] = 0u;
  g1.w[7] = 0u;
  v4i z4 = {0, 0, 0, 0};
#if __clang_major__ >= 23
  v8i z8 = {0, 0, 0, 0, 0, 0, 0, 0};
  __builtin_amdgcn_tensor_load_to_lds(g0, g1.v, z4, z4, z8, 0);
#else
  __builtin_amdgcn_tensor_load_to_lds(g0, g1.v, z4, z4, 0);
#endif
}
#endif

// ---- elementwise prep kernels ----------------------------------------------
__global__ void cast_bf16_kernel(const float* in, __bf16* out, int n) {
  int i = blockIdx.x * 256 + threadIdx.x;
  if (i < n) out[i] = (__bf16)in[i];
}
__global__ void cast_transpose_kernel(const float* W, __bf16* Wt, int K, int N) {
  int i = blockIdx.x * 256 + threadIdx.x;
  if (i >= K * N) return;
  int k = i / N, c = i % N;
  Wt[(size_t)c * K + k] = (__bf16)W[i];
}
__global__ void transpose_v_kernel(const __bf16* v, __bf16* vT) {
  int i = blockIdx.x * 256 + threadIdx.x;
  if (i >= MROWS * DMODEL) return;
  int bt = i >> 10, dd = i & 1023;
  int b = bt >> 10, t = bt & 1023;
  int hh = dd >> 6, e = dd & 63;
  vT[((size_t)(b * NHEAD + hh) * HDIM + e) * NTOK + t] = v[i];
}
__global__ void small_gemm_kernel(const float* A, const float* B, float* out, int M, int N, int K) {
  int i = blockIdx.x * 256 + threadIdx.x;
  if (i >= M * N) return;
  int m = i / N, n = i % N;
  float s = 0.f;
  for (int k = 0; k < K; ++k) s += A[(size_t)m * K + k] * B[(size_t)k * N + n];
  out[i] = s;
}
__global__ void prep_kernel(const float* wraw, const float* betaPre, const float* fPre,
                            const float* delta, __bf16* wbf, float* betaT, float* logf_out) {
  __shared__ float hsum[NHEAD];
  int bt = blockIdx.x;
  int b = bt >> 10, t = bt & 1023;
  const float* wr = wraw + (size_t)bt * DMODEL;
  if (threadIdx.x < NHEAD) hsum[threadIdx.x] = 0.f;
  __syncthreads();
  int e0 = threadIdx.x * 4;
  float4 wv = *(const float4*)(wr + e0);
  atomicAdd(&hsum[e0 >> 6], wv.x * wv.x + wv.y * wv.y + wv.z * wv.z + wv.w * wv.w);
  __syncthreads();
  float rn = rsqrtf(hsum[e0 >> 6] + 1e-6f);
  __bf16* wo = wbf + (size_t)bt * DMODEL;
  wo[e0 + 0] = (__bf16)(wv.x * rn);
  wo[e0 + 1] = (__bf16)(wv.y * rn);
  wo[e0 + 2] = (__bf16)(wv.z * rn);
  wo[e0 + 3] = (__bf16)(wv.w * rn);
  if (threadIdx.x < NHEAD) {
    int hh = threadIdx.x;
    float bp = betaPre[(size_t)bt * NHEAD + hh];
    betaT[((size_t)(b * NHEAD + hh)) * NTOK + t] = 2.0f / (1.0f + __expf(-bp));
    float fp = fPre[(size_t)bt * NHEAD + hh] + delta[hh];
    logf_out[(size_t)bt * NHEAD + hh] = fminf(fp, 0.0f) - log1pf(__expf(-fabsf(fp)));
  }
}
__global__ __launch_bounds__(1024) void cumsum_kernel(const float* logf, float* FcT) {
  __shared__ float buf[NTOK];
  int bh = blockIdx.x, b = bh >> 4, hh = bh & 15;
  int t = threadIdx.x;
  buf[t] = logf[((size_t)b * NTOK + t) * NHEAD + hh];
  __syncthreads();
  for (int off = 1; off < NTOK; off <<= 1) {
    float v = (t >= off) ? buf[t - off] : 0.0f;
    __syncthreads();
    buf[t] += v;
    __syncthreads();
  }
  FcT[(size_t)bh * NTOK + t] = buf[t];
}

// ---- big WMMA GEMM with TDM-staged, double-buffered LDS tiles ---------------
// out(M,N) = A(M,K) @ Bt(N,K)^T ; 128x128 block tile, 8 waves, 32-deep k steps.
// LDS tile rows are 64B data + 16B pad -> 80B stride: conflict-free b128 reads.
__global__ __launch_bounds__(256) void gemm_bf16_kernel(const __bf16* A, const __bf16* Bt,
                                                        __bf16* outBf, float* outF,
                                                        int M, int N, int K) {
  __shared__ alignas(16) char As[2][128 * 80];
  __shared__ alignas(16) char Bs[2][128 * 80];
  int wave = threadIdx.x >> 5, lane = threadIdx.x & 31;
  int cl = lane & 15, hf = lane >> 4;
  int r0 = blockIdx.y * 128 + wave * 16;
  int c0 = blockIdx.x * 128;
  const __bf16* Ag = A + (size_t)(blockIdx.y * 128) * K;
  const __bf16* Bg = Bt + (size_t)c0 * K;

  auto fill = [&](int buf, int kb) {
#if HAVE_TDM
    if (wave == 0) {
      tdm_issue_2d((unsigned)(size_t)&As[buf][0], Ag + kb, K);
      tdm_issue_2d((unsigned)(size_t)&Bs[buf][0], Bg + kb, K);
    }
#else
    for (int c = (int)threadIdx.x; c < 512; c += 256) {
      int row = c >> 2, kc = (c & 3) * 8;
      *(uint4*)(As[buf] + row * 80 + kc * 2) = *(const uint4*)(Ag + (size_t)row * K + kb + kc);
      *(uint4*)(Bs[buf] + row * 80 + kc * 2) = *(const uint4*)(Bg + (size_t)row * K + kb + kc);
    }
#endif
  };
  auto fence = [&]() {
#if HAVE_TDM
    if (wave == 0) __builtin_amdgcn_s_wait_tensorcnt((short)0);
#endif
    __syncthreads();
  };

  v8f acc[8] = {};
  fill(0, 0);
  fence();
  int buf = 0;
  for (int kb = 0; kb < K; kb += 32, buf ^= 1) {
    if (kb + 32 < K) fill(buf ^ 1, kb + 32);
    const char* Ab = As[buf];
    const char* Bb = Bs[buf];
    int arow = wave * 16 + cl;
    v16bf a;
    ((uint4*)&a)[0] = *(const uint4*)(Ab + arow * 80 + hf * 16);
    ((uint4*)&a)[1] = *(const uint4*)(Ab + arow * 80 + 32 + hf * 16);
    v16bf bb[8];
#pragma unroll
    for (int cf = 0; cf < 8; ++cf) {
      int brow = cf * 16 + cl;
      ((uint4*)&bb[cf])[0] = *(const uint4*)(Bb + brow * 80 + hf * 16);
      ((uint4*)&bb[cf])[1] = *(const uint4*)(Bb + brow * 80 + 32 + hf * 16);
    }
#pragma unroll
    for (int cf = 0; cf < 8; ++cf) acc[cf] = WMMA_BF16(a, bb[cf], acc[cf]);
    fence();
  }
#pragma unroll
  for (int cf = 0; cf < 8; ++cf) {
    int col = c0 + cf * 16 + cl;
#pragma unroll
    for (int i = 0; i < 8; ++i) {
      int row = r0 + hf * 8 + i;
      size_t o = (size_t)row * N + col;
      if (outBf) outBf[o] = (__bf16)acc[cf][i];
      if (outF)  outF[o]  = acc[cf][i];
    }
  }
}

// ---- triangular per-(b,h) gram GEMM: T = Arows @ Brows^T over hd ------------
__global__ __launch_bounds__(256) void tri_gemm_kernel(const __bf16* Asrc, const __bf16* Bsrc,
                                                       const float* colScale,
                                                       __bf16* outBf, float* outF, int strict) {
  int bh = blockIdx.x, b = bh >> 4, hh = bh & 15;
  int idx = blockIdx.y, ti = 0, rem = idx;
  while (rem > ti) { rem -= ti + 1; ++ti; }
  int tj = rem;
  int wave = threadIdx.x >> 5, lane = threadIdx.x & 31;
  int cl = lane & 15, hf = lane >> 4;
  int t0 = ti * 128 + wave * 16, c0 = tj * 128;
  const __bf16* Ab = Asrc + (size_t)b * NTOK * DMODEL + hh * HDIM;
  const __bf16* Bb = Bsrc + (size_t)b * NTOK * DMODEL + hh * HDIM;
  v8f acc[8] = {};
  for (int kb = 0; kb < HDIM; kb += 32) {
    v16bf a = load_a_frag(Ab + (size_t)t0 * DMODEL, DMODEL, kb);
    v16bf bb[8];
#pragma unroll
    for (int cf = 0; cf < 8; ++cf) bb[cf] = load_b_frag(Bb, DMODEL, c0 + cf * 16, kb);
#pragma unroll
    for (int cf = 0; cf < 8; ++cf) acc[cf] = WMMA_BF16(a, bb[cf], acc[cf]);
  }
  size_t obase = (size_t)bh * NTOK * NTOK;
#pragma unroll
  for (int cf = 0; cf < 8; ++cf) {
    int col = c0 + cf * 16 + cl;
    float sc = colScale ? colScale[(size_t)bh * NTOK + col] : 1.0f;
#pragma unroll
    for (int i = 0; i < 8; ++i) {
      int row = t0 + hf * 8 + i;
      bool keep = strict ? (row > col) : (row >= col);
      float v = keep ? acc[cf][i] * sc : 0.0f;
      if (outBf) outBf[obase + (size_t)row * NTOK + col] = (__bf16)v;
      else       outF[obase + (size_t)row * NTOK + col]  = v;
    }
  }
}

// ---- blocked unit-lower triangular solve (I+L)C = Mlow, C stored transposed -
__global__ __launch_bounds__(256) void solve_kernel(const __bf16* Lbf, const float* Mlow, __bf16* CbfT) {
  __shared__ float Us[8][16][16];  // [wave][col][row]
  int bh = blockIdx.x, jt = blockIdx.y;
  size_t base = (size_t)bh * NTOK * NTOK;
  const __bf16* L = Lbf + base;
  const float*  M = Mlow + base;
  __bf16* Ct = CbfT + base;
  int wave = threadIdx.x >> 5, lane = threadIdx.x & 31;
  int cl = lane & 15, hf = lane >> 4;
  int c0 = jt * 128 + wave * 16;
  {
    int zc = c0 + (lane >> 1);
    int zs = (lane & 1) * 8;
    uint4 z = {0u, 0u, 0u, 0u};
    __bf16* zp = Ct + (size_t)zc * NTOK;
    for (int r = zs; r < NTOK; r += 16) *(uint4*)(zp + r) = z;
  }
  asm volatile("s_wait_storecnt 0" ::: "memory");
  __syncthreads();

  int rb0 = jt * 8;
  for (int rb = rb0; rb < NTOK / 16; ++rb) {
    int s0 = rb * 16;
    v8f acc;
#pragma unroll
    for (int i = 0; i < 8; ++i)
      acc[i] = M[(size_t)(s0 + hf * 8 + i) * NTOK + c0 + cl];
    for (int kb = jt * 128; kb < s0; kb += 32) {
      v16bf a = neg_frag(load_a_frag(L + (size_t)s0 * NTOK, NTOK, kb));
      v16bf b = load_b_frag(Ct, NTOK, c0, kb);
      acc = WMMA_BF16(a, b, acc);
    }
#pragma unroll
    for (int i = 0; i < 8; ++i) Us[wave][cl][hf * 8 + i] = acc[i];
    __syncthreads();
    if (lane < 16) {
      int colg = c0 + lane;
      float c[16];
#pragma unroll
      for (int i = 0; i < 16; ++i) {
        float u = Us[wave][lane][i];
#pragma unroll
        for (int j = 0; j < 16; ++j)
          if (j < i) u -= (float)L[(size_t)(s0 + i) * NTOK + s0 + j] * c[j];
        c[i] = u;
      }
#pragma unroll
      for (int i = 0; i < 16; ++i) Ct[(size_t)colg * NTOK + s0 + i] = (__bf16)c[i];
    }
    asm volatile("s_wait_storecnt 0" ::: "memory");
    __syncthreads();
  }
}

// ---- fused logits: S = QK^T*scale - (G@C)*scale + Fc_t - Fc_j, causal mask --
__global__ __launch_bounds__(256) void s_kernel(const __bf16* Gbf, const __bf16* CbfT,
                                                const __bf16* qbf, const __bf16* kbf,
                                                const float* FcT, float* Sl) {
  int bh = blockIdx.x, b = bh >> 4, hh = bh & 15;
  int idx = blockIdx.y, ti = 0, rem = idx;
  while (rem > ti) { rem -= ti + 1; ++ti; }
  int tj = rem;
  int wave = threadIdx.x >> 5, lane = threadIdx.x & 31;
  int cl = lane & 15, hf = lane >> 4;
  int t0 = ti * 128 + wave * 16, c0 = tj * 128;
  const __bf16* Gb = Gbf + (size_t)bh * NTOK * NTOK;
  const __bf16* Ct = CbfT + (size_t)bh * NTOK * NTOK;
  v8f acc[8] = {};
  for (int kb = tj * 128; kb < (ti + 1) * 128; kb += 32) {
    __builtin_prefetch(Gb + (size_t)t0 * NTOK + kb + 64, 0, 1);
    v16bf a = neg_frag(load_a_frag(Gb + (size_t)t0 * NTOK, NTOK, kb));
    v16bf bb[8];
#pragma unroll
    for (int cf = 0; cf < 8; ++cf) bb[cf] = load_b_frag(Ct, NTOK, c0 + cf * 16, kb);
#pragma unroll
    for (int cf = 0; cf < 8; ++cf) acc[cf] = WMMA_BF16(a, bb[cf], acc[cf]);
  }
  const __bf16* Qb = qbf + (size_t)b * NTOK * DMODEL + hh * HDIM;
  const __bf16* Kb = kbf + (size_t)b * NTOK * DMODEL + hh * HDIM;
  for (int kb = 0; kb < HDIM; kb += 32) {
    v16bf a = load_a_frag(Qb + (size_t)t0 * DMODEL, DMODEL, kb);
    v16bf bb[8];
#pragma unroll
    for (int cf = 0; cf < 8; ++cf) bb[cf] = load_b_frag(Kb, DMODEL, c0 + cf * 16, kb);
#pragma unroll
    for (int cf = 0; cf < 8; ++cf) acc[cf] = WMMA_BF16(a, bb[cf], acc[cf]);
  }
  const float* Fc = FcT + (size_t)bh * NTOK;
  size_t obase = (size_t)bh * NTOK * NTOK;
#pragma unroll
  for (int cf = 0; cf < 8; ++cf) {
    int col = c0 + cf * 16 + cl;
    float fcj = Fc[col];
#pragma unroll
    for (int i = 0; i < 8; ++i) {
      int row = t0 + hf * 8 + i;
      float v = acc[cf][i] * ATT_SCALE + Fc[row] - fcj;
      if (col > row) v = -1e30f;
      Sl[obase + (size_t)row * NTOK + col] = v;
    }
  }
}

// ---- softmax: one wave per row ----------------------------------------------
__global__ __launch_bounds__(256) void softmax_kernel(const float* Sl, __bf16* Pbf) {
  int wid = blockIdx.x * 8 + (threadIdx.x >> 5);
  int lane = threadIdx.x & 31;
  int bh = wid >> 10, t = wid & 1023;
  const float* row = Sl + ((size_t)bh * NTOK + t) * NTOK;
  float m = -3.0e38f;
  for (int j = lane; j <= t; j += 32) m = fmaxf(m, row[j]);
  for (int off = 16; off; off >>= 1) m = fmaxf(m, __shfl_xor(m, off, 32));
  float s = 0.f;
  for (int j = lane; j <= t; j += 32) s += __expf(row[j] - m);
  for (int off = 16; off; off >>= 1) s += __shfl_xor(s, off, 32);
  float inv = 1.0f / s;
  __bf16* prow = Pbf + ((size_t)bh * NTOK + t) * NTOK;
  int rend = (t | 127) + 1;
  for (int j = lane; j < rend; j += 32)
    prow[j] = (__bf16)((j <= t) ? __expf(row[j] - m) * inv : 0.0f);
}

// ---- o = P @ V per (b,h) ----------------------------------------------------
__global__ __launch_bounds__(256) void pv_kernel(const __bf16* Pbf, const __bf16* vT, __bf16* obf) {
  int bh = blockIdx.x, ti = blockIdx.y;
  int b = bh >> 4, hh = bh & 15;
  int wave = threadIdx.x >> 5, lane = threadIdx.x & 31;
  int cl = lane & 15, hf = lane >> 4;
  int t0 = ti * 128 + wave * 16;
  const __bf16* Pb = Pbf + (size_t)bh * NTOK * NTOK;
  const __bf16* Vt = vT + (size_t)bh * HDIM * NTOK;
  v8f acc[4] = {};
  for (int kb = 0; kb < (ti + 1) * 128; kb += 32) {
    __builtin_prefetch(Pb + (size_t)t0 * NTOK + kb + 64, 0, 1);
    v16bf a = load_a_frag(Pb + (size_t)t0 * NTOK, NTOK, kb);
    v16bf bb[4];
#pragma unroll
    for (int cf = 0; cf < 4; ++cf) bb[cf] = load_b_frag(Vt, NTOK, cf * 16, kb);
#pragma unroll
    for (int cf = 0; cf < 4; ++cf) acc[cf] = WMMA_BF16(a, bb[cf], acc[cf]);
  }
#pragma unroll
  for (int cf = 0; cf < 4; ++cf) {
    int e = cf * 16 + cl;
#pragma unroll
    for (int i = 0; i < 8; ++i) {
      int row = t0 + hf * 8 + i;
      obf[(size_t)(b * NTOK + row) * DMODEL + hh * HDIM + e] = (__bf16)acc[cf][i];
    }
  }
}

// ---- host launch ------------------------------------------------------------
extern "C" void kernel_launch(void* const* d_in, const int* in_sizes, int n_in,
                              void* d_out, int out_size, void* d_ws, size_t ws_size,
                              hipStream_t stream) {
  (void)in_sizes; (void)n_in; (void)out_size; (void)ws_size;
  const float* x   = (const float*)d_in[0];
  const float* Wq  = (const float*)d_in[1];
  const float* Wk  = (const float*)d_in[2];
  const float* Wv  = (const float*)d_in[3];
  const float* Wo  = (const float*)d_in[4];
  const float* Ww1 = (const float*)d_in[5];
  const float* Ww2 = (const float*)d_in[6];
  const float* Wb  = (const float*)d_in[7];
  const float* Wf  = (const float*)d_in[8];
  const float* dl  = (const float*)d_in[9];
  float* out = (float*)d_out;

  size_t off = 0;
  char* ws = (char*)d_ws;
  auto take = [&](size_t bytes) -> char* {
    char* p = ws + off;
    off = (off + bytes + 255) & ~(size_t)255;
    return p;
  };
  const size_t MD = (size_t)MROWS * DMODEL;
  const size_t DD = (size_t)DMODEL * DMODEL;
  const size_t NN2 = (size_t)NBH * NTOK * NTOK;

  __bf16* xbf  = (__bf16*)take(MD * 2);
  __bf16* WqT  = (__bf16*)take(DD * 2);
  __bf16* WkT  = (__bf16*)take(DD * 2);
  __bf16* WvT  = (__bf16*)take(DD * 2);
  __bf16* WoT  = (__bf16*)take(DD * 2);
  __bf16* qbf  = (__bf16*)take(MD * 2);
  __bf16* kbf  = (__bf16*)take(MD * 2);
  __bf16* vbf  = (__bf16*)take(MD * 2);
  __bf16* wbf  = (__bf16*)take(MD * 2);
  __bf16* obf  = (__bf16*)take(MD * 2);
  float*  t1   = (float*)take((size_t)MROWS * HDIM * 4);
  float*  wraw = (float*)take(MD * 4);
  float*  betaPre = (float*)take((size_t)MROWS * NHEAD * 4);
  float*  fPre    = (float*)take((size_t)MROWS * NHEAD * 4);
  float*  betaT   = (float*)take((size_t)NBH * NTOK * 4);
  float*  logf    = (float*)take((size_t)MROWS * NHEAD * 4);
  float*  FcT     = (float*)take((size_t)NBH * NTOK * 4);
  __bf16* vT   = (__bf16*)take((size_t)NBH * HDIM * NTOK * 2);
  __bf16* Lbf  = (__bf16*)take(NN2 * 2);
  float*  Mlow = (float*)take(NN2 * 4);
  __bf16* CbfT = (__bf16*)take(NN2 * 2);
  __bf16* Gbf  = (__bf16*)take(NN2 * 2);
  float*  Sl   = (float*)take(NN2 * 4);
  __bf16* Pbf  = (__bf16*)take(NN2 * 2);

  cast_bf16_kernel<<<(int)(MD / 256), 256, 0, stream>>>(x, xbf, (int)MD);
  cast_transpose_kernel<<<(int)(DD / 256), 256, 0, stream>>>(Wq, WqT, DMODEL, DMODEL);
  cast_transpose_kernel<<<(int)(DD / 256), 256, 0, stream>>>(Wk, WkT, DMODEL, DMODEL);
  cast_transpose_kernel<<<(int)(DD / 256), 256, 0, stream>>>(Wv, WvT, DMODEL, DMODEL);
  cast_transpose_kernel<<<(int)(DD / 256), 256, 0, stream>>>(Wo, WoT, DMODEL, DMODEL);

  dim3 gproj(DMODEL / 128, MROWS / 128);
  gemm_bf16_kernel<<<gproj, 256, 0, stream>>>(xbf, WqT, qbf, nullptr, MROWS, DMODEL, DMODEL);
  gemm_bf16_kernel<<<gproj, 256, 0, stream>>>(xbf, WkT, kbf, nullptr, MROWS, DMODEL, DMODEL);
  gemm_bf16_kernel<<<gproj, 256, 0, stream>>>(xbf, WvT, vbf, nullptr, MROWS, DMODEL, DMODEL);

  small_gemm_kernel<<<(MROWS * NHEAD) / 256, 256, 0, stream>>>(x, Wb, betaPre, MROWS, NHEAD, DMODEL);
  small_gemm_kernel<<<(MROWS * NHEAD) / 256, 256, 0, stream>>>(x, Wf, fPre, MROWS, NHEAD, DMODEL);
  small_gemm_kernel<<<(MROWS * HDIM) / 256, 256, 0, stream>>>(x, Ww1, t1, MROWS, HDIM, DMODEL);
  small_gemm_kernel<<<(int)(MD / 256), 256, 0, stream>>>(t1, Ww2, wraw, MROWS, DMODEL, HDIM);
  prep_kernel<<<MROWS, 256, 0, stream>>>(wraw, betaPre, fPre, dl, wbf, betaT, logf);
  cumsum_kernel<<<NBH, 1024, 0, stream>>>(logf, FcT);
  transpose_v_kernel<<<(int)(MD / 256), 256, 0, stream>>>(vbf, vT);

  dim3 gtri(NBH, 36);
  tri_gemm_kernel<<<gtri, 256, 0, stream>>>(wbf, wbf, betaT, Lbf, nullptr, 1);
  tri_gemm_kernel<<<gtri, 256, 0, stream>>>(wbf, kbf, nullptr, nullptr, Mlow, 1);
  tri_gemm_kernel<<<gtri, 256, 0, stream>>>(qbf, wbf, betaT, Gbf, nullptr, 0);

  solve_kernel<<<dim3(NBH, NTOK / 128), 256, 0, stream>>>(Lbf, Mlow, CbfT);
  s_kernel<<<gtri, 256, 0, stream>>>(Gbf, CbfT, qbf, kbf, FcT, Sl);
  softmax_kernel<<<(NBH * NTOK) / 8, 256, 0, stream>>>(Sl, Pbf);
  pv_kernel<<<dim3(NBH, NTOK / 128), 256, 0, stream>>>(Pbf, vT, obf);
  gemm_bf16_kernel<<<gproj, 256, 0, stream>>>(obf, WoT, nullptr, out, MROWS, DMODEL, DMODEL);
}
